// attention_76416058130792
// MI455X (gfx1250) — matrix operand order repaired and checked by reading, not compiled, not run on hardware
//
#include <hip/hip_runtime.h>
#include <hip/hip_bf16.h>

typedef _Float16 half_t;
typedef __attribute__((ext_vector_type(16))) _Float16 v16h;
typedef __attribute__((ext_vector_type(8)))  _Float16 v8h;
typedef __attribute__((ext_vector_type(8)))  float    v8f;

#define ATT_B 4
#define ATT_L 4096
#define ATT_D 64

// 1/sqrt(64) * log2(e): folded into the f16 copy of Q so WMMA scores come out
// directly in the log2 domain (softmax is invariant under the base change,
// and v_exp_f32 / v_log_f32 are natively base-2 on AMD hardware).
#define ATT_QSCALE 0.1803368801111204f

__device__ __forceinline__ v8f wmma_f16(v16h a, v16h b, v8f c) {
    // D = A(16x32 f16) * B(32x16 f16) + C(16x16 f32)
    return __builtin_amdgcn_wmma_f32_16x16x32_f16(
        /*neg_a=*/false, a, /*neg_b=*/false, b,
        /*c_mod=*/(short)0, c, /*reuse_a=*/false, /*reuse_b=*/false);
}

__device__ __forceinline__ float fast_exp2(float x) { return __builtin_amdgcn_exp2f(x); }
__device__ __forceinline__ float fast_log2(float x) { return __builtin_amdgcn_logf(x); }

// Load a 16x32 f16 A-fragment from a row-major [rows][64] f16 matrix.
// Layout: lane (m = lane&15, g = lane>>4) holds K run [16g, 16g+16)
// contiguously (2 packed halfs per VGPR), matching the B fragments.
__device__ __forceinline__ v16h load_a_frag(const half_t* src, int row0, int dk, int lane) {
    int m = lane & 15;
    int g = lane >> 4;
    const half_t* p = src + (size_t)(row0 + m) * ATT_D + dk;
    v8h lo = *(const v8h*)(p + 16 * g);       // K = 16g .. 16g+7
    v8h hi = *(const v8h*)(p + 16 * g + 8);   // K = 16g+8 .. 16g+15
    v16h r;
#pragma unroll
    for (int t = 0; t < 8; ++t) { r[t] = lo[t]; r[8 + t] = hi[t]; }
    return r;
}

// ---------------------------------------------------------------------------
// Kernel 0: fp32 -> f16 conversion.  qh is PRE-SCALED by scale*log2(e);
// kh row-major; vt transposed [b][d][i] so V B-fragments are contiguous.
// ---------------------------------------------------------------------------
__global__ void __launch_bounds__(256)
attn_convert_kernel(const float* __restrict__ q, const float* __restrict__ k,
                    const float* __restrict__ v,
                    half_t* __restrict__ qh, half_t* __restrict__ kh,
                    half_t* __restrict__ vt) {
    size_t idx = (size_t)blockIdx.x * blockDim.x + threadIdx.x;
    size_t n = (size_t)ATT_B * ATT_L * ATT_D;
    if (idx >= n) return;
    qh[idx] = (half_t)(q[idx] * ATT_QSCALE);
    kh[idx] = (half_t)k[idx];
    int d = (int)(idx % ATT_D);
    int i = (int)((idx / ATT_D) % ATT_L);
    int b = (int)(idx / ((size_t)ATT_D * ATT_L));
    vt[((size_t)b * ATT_D + d) * ATT_L + i] = (half_t)v[idx];
}

// ---------------------------------------------------------------------------
// Kernel 1: per-row log2-sum-exp2 of (pre-scaled) scores.  One wave owns 16
// query rows and streams keys in 64-wide j tiles (8 WMMAs per iteration).
// Per row: tile max via cheap v_max first, then ONE rescale exp2 + 4 element
// exp2s (1.25 trans/element), 4 independent TRANS streams to co-issue with
// the WMMAs.  Butterfly reduction across the 16-lane C-matrix halves; store
// lse2[b*L+i] = m + log2(sum).
// ---------------------------------------------------------------------------
__global__ void __launch_bounds__(128)
attn_rowstats_kernel(const half_t* __restrict__ qh, const half_t* __restrict__ kh,
                     float* __restrict__ lse) {
    const int wave = threadIdx.x >> 5;
    const int lane = threadIdx.x & 31;
    const int b    = blockIdx.y;
    const int i0   = blockIdx.x * 64 + wave * 16;
    const half_t* qb = qh + (size_t)b * ATT_L * ATT_D;
    const half_t* kb = kh + (size_t)b * ATT_L * ATT_D;

    const v16h a0 = load_a_frag(qb, i0, 0,  lane);  // Q rows, d = 0..31
    const v16h a1 = load_a_frag(qb, i0, 32, lane);  // Q rows, d = 32..63

    float mx[8], sm[8];
#pragma unroll
    for (int r = 0; r < 8; ++r) { mx[r] = -3.0e38f; sm[r] = 0.0f; }

    const int n = lane & 15;
    const int h = lane >> 4;
    for (int j0 = 0; j0 < ATT_L; j0 += 64) {
        v8f c[4];
        // 4 sub-tiles of 16 j columns each: B = K^T, lane holds column
        // j0+16t+n with 16 contiguous d values.
#pragma unroll
        for (int t = 0; t < 4; ++t) {
            const half_t* kp = kb + (size_t)(j0 + 16 * t + n) * ATT_D + 16 * h;
            v16h b0 = *(const v16h*)(kp);        // d = 0..31 half
            v16h b1 = *(const v16h*)(kp + 32);   // d = 32..63 half
            v8f acc = {};
            acc = wmma_f16(a0, b0, acc);
            acc = wmma_f16(a1, b1, acc);
            c[t] = acc;
        }
        __builtin_prefetch(kb + (size_t)(j0 + 64 + n) * ATT_D, 0, 3);
#pragma unroll
        for (int r = 0; r < 8; ++r) {
            float s0 = c[0][r];
            float s1 = c[1][r];
            float s2 = c[2][r];
            float s3 = c[3][r];
            float tm = fmaxf(fmaxf(s0, s1), fmaxf(s2, s3));
            float nm = fmaxf(mx[r], tm);
            float e0 = fast_exp2(s0 - nm);
            float e1 = fast_exp2(s1 - nm);
            float e2 = fast_exp2(s2 - nm);
            float e3 = fast_exp2(s3 - nm);
            sm[r] = sm[r] * fast_exp2(mx[r] - nm) + ((e0 + e1) + (e2 + e3));
            mx[r] = nm;
        }
    }
    // Reduce across the 16 lanes of each half (rows r / 8+r).
#pragma unroll
    for (int mask = 1; mask <= 8; mask <<= 1) {
#pragma unroll
        for (int r = 0; r < 8; ++r) {
            float mo = __shfl_xor(mx[r], mask, 32);
            float so = __shfl_xor(sm[r], mask, 32);
            float nm = fmaxf(mx[r], mo);
            sm[r] = sm[r] * fast_exp2(mx[r] - nm) + so * fast_exp2(mo - nm);
            mx[r] = nm;
        }
    }
    if (n == 0) {
        int rbase = i0 + h * 8;
#pragma unroll
        for (int r = 0; r < 8; ++r)
            lse[(size_t)b * ATT_L + rbase + r] = mx[r] + fast_log2(sm[r]);
    }
}

// ---------------------------------------------------------------------------
// Kernel 2: out[b][j][d] = sum_i exp2(s2_ij - lse2_i) * v[b][i][d]  (scores
// already in log2 domain via pre-scaled Q).  One wave owns a 16-wide j tile;
// computes S^T = K_j @ Q_i^T so lse2_i is a per-lane constant, transposes P^T
// through per-wave LDS into an A-fragment, then accumulates O += P^T @ V via
// WMMA (4 d-tiles).
// ---------------------------------------------------------------------------
__global__ void __launch_bounds__(128)
attn_out_kernel(const half_t* __restrict__ qh, const half_t* __restrict__ kh,
                const half_t* __restrict__ vt, const float* __restrict__ lse,
                float* __restrict__ out) {
    __shared__ __align__(32) half_t lds[4][16 * 32];  // per-wave P^T [j][i]
    const int wave = threadIdx.x >> 5;
    const int lane = threadIdx.x & 31;
    const int b    = blockIdx.y;
    const int j0   = blockIdx.x * 64 + wave * 16;
    const half_t* qb  = qh + (size_t)b * ATT_L * ATT_D;
    const half_t* kb  = kh + (size_t)b * ATT_L * ATT_D;
    const half_t* vtb = vt + (size_t)b * ATT_D * ATT_L;
    const float*  lb  = lse + (size_t)b * ATT_L;

    // A fragments of K rows (constant across the i loop).
    const v16h ak0 = load_a_frag(kb, j0, 0,  lane);
    const v16h ak1 = load_a_frag(kb, j0, 32, lane);

    const int n = lane & 15;
    const int h = lane >> 4;
    half_t* P = &lds[wave][0];

    v8f acc[4] = {{}, {}, {}, {}};

    for (int i0 = 0; i0 < ATT_L; i0 += 32) {
        __builtin_prefetch(qb + (size_t)(i0 + 32 + n) * ATT_D, 0, 3);
        // ---- S^T tiles (j x i), two 16-wide i sub-tiles ----
#pragma unroll
        for (int t = 0; t < 2; ++t) {
            const int irow = i0 + 16 * t + n;  // this lane's i index (column)
            const half_t* qp = qb + (size_t)irow * ATT_D + 16 * h;
            v16h bq0 = *(const v16h*)(qp);        // B = Q^T, d = 0..31
            v16h bq1 = *(const v16h*)(qp + 32);   // B = Q^T, d = 32..63
            v8f st = {};
            st = wmma_f16(ak0, bq0, st);
            st = wmma_f16(ak1, bq1, st);
            const float l = lb[irow];             // per-lane constant (per i)
            // C layout: row j = j0 + r + 8h, col i = irow.  Store P^T[j][i].
#pragma unroll
            for (int r = 0; r < 8; ++r) {
                float p = fast_exp2(st[r] - l);
                P[(r + 8 * h) * 32 + 16 * t + n] = (half_t)p;
            }
        }
        // ---- Re-load P^T as a 16x32 A-fragment (same-wave LDS, in order) ----
        const half_t* pp = P + n * 32;            // row j = j0 + n
        v8h lo = *(const v8h*)(pp + 16 * h);      // K(i) = 16h .. 16h+7
        v8h hi = *(const v8h*)(pp + 16 * h + 8);  // K(i) = 16h+8 .. +7
        v16h ap;
#pragma unroll
        for (int t = 0; t < 8; ++t) { ap[t] = lo[t]; ap[8 + t] = hi[t]; }
        // ---- O += P^T @ V over four 16-wide d tiles ----
#pragma unroll
        for (int dt = 0; dt < 4; ++dt) {
            v16h bv = *(const v16h*)(vtb + (size_t)(dt * 16 + n) * ATT_L + i0 + 16 * h);
            acc[dt] = wmma_f16(ap, bv, acc[dt]);
        }
    }

    // Store: C layout row j = j0 + r + 8h, col d = dt*16 + n.
#pragma unroll
    for (int dt = 0; dt < 4; ++dt) {
#pragma unroll
        for (int r = 0; r < 8; ++r) {
            int j = j0 + r + 8 * h;
            out[((size_t)b * ATT_L + j) * ATT_D + dt * 16 + n] = acc[dt][r];
        }
    }
}

// ---------------------------------------------------------------------------
extern "C" void kernel_launch(void* const* d_in, const int* in_sizes, int n_in,
                              void* d_out, int out_size, void* d_ws, size_t ws_size,
                              hipStream_t stream) {
    const float* q = (const float*)d_in[0];
    const float* k = (const float*)d_in[1];
    const float* v = (const float*)d_in[2];
    float* out = (float*)d_out;

    const size_t nElem = (size_t)ATT_B * ATT_L * ATT_D;  // 1,048,576
    half_t* qh = (half_t*)d_ws;
    half_t* kh = qh + nElem;
    half_t* vt = kh + nElem;
    float*  lse = (float*)(vt + nElem);                  // B*L floats

    attn_convert_kernel<<<(unsigned)((nElem + 255) / 256), 256, 0, stream>>>(
        q, k, v, qh, kh, vt);

    dim3 grid(ATT_L / 64, ATT_B);   // 4 waves/block, 16 rows (cols) per wave
    attn_rowstats_kernel<<<grid, 128, 0, stream>>>(qh, kh, lse);
    attn_out_kernel<<<grid, 128, 0, stream>>>(qh, kh, vt, lse, out);
}
